// MessagePassing_36283883716812
// MI455X (gfx1250) — matrix operand order repaired and checked by reading, not compiled, not run on hardware
//
#include <hip/hip_runtime.h>

// Problem constants (from the reference)
#define BB 8
#define HH 128
#define WW 256
#define CC 128
#define KK 9
#define PAD 4

// WMMA vector types (probe-confirmed signatures)
typedef __attribute__((ext_vector_type(16))) __bf16 v16bf;
typedef __attribute__((ext_vector_type(8)))  float  v8f;

// TDM builtin arg vector types (6-arg toolchain: clang-23 / therock-10.0)
typedef __attribute__((ext_vector_type(4))) unsigned int uint4v;
typedef __attribute__((ext_vector_type(8))) int          int8v;
typedef __attribute__((ext_vector_type(4))) int          int4v;

// Weights in bf16 B-fragment layout:
//   frag f = ((tap*4 + kc)*8 + ntile), 288 frags, 1 KB each (32 lanes x 16 bf16)
//   within frag: lane L holds N = L%16, half = L/16; element e -> K = half*16 + e
#define WFRAGS   (KK * 4 * 8)            // 288
#define WBYTES   (WFRAGS * 1024)         // 294912 bytes (288 KB) -> fits 320 KB LDS

__device__ __forceinline__ unsigned short f32_to_bf16_rne(float f) {
    unsigned int u = __float_as_uint(f);
    unsigned int r = u + 0x7FFFu + ((u >> 16) & 1u);   // round-to-nearest-even
    return (unsigned short)(r >> 16);
}

// Pack two fp32 -> dword of two bf16 (truncation) with a single v_perm_b32.
// D.b[0..1] = lo.b[2..3], D.b[2..3] = hi.b[2..3]  (sel: 0-3 -> src1, 4-7 -> src0)
__device__ __forceinline__ unsigned int pack_bf16_trunc(float lo, float hi) {
    return __builtin_amdgcn_perm(__float_as_uint(hi), __float_as_uint(lo),
                                 0x07060302u);
}

// ---------------------------------------------------------------------------
// Kernel 0: convert fp32 conv weights (K, Cin, Cout) into bf16 B-fragments
// ---------------------------------------------------------------------------
__global__ void wcvt_kernel(const float* __restrict__ Wf,
                            unsigned short* __restrict__ wsbf) {
    int idx = blockIdx.x * 256 + threadIdx.x;          // < 147456
    int f   = idx >> 9;                                // fragment id
    int r   = idx & 511;                               // element within fragment
    int L   = r >> 4;                                  // lane 0..31
    int e   = r & 15;                                  // element 0..15
    int nt  = f & 7;
    int kc  = (f >> 3) & 3;
    int tap = f >> 5;
    int hf  = L >> 4;
    int Kk  = hf * 16 + e;                             // K within 32-chunk
    int ci  = kc * 32 + Kk;
    int co  = nt * 16 + (L & 15);
    wsbf[idx] = f32_to_bf16_rne(Wf[(tap * CC + ci) * CC + co]);
}

// ---------------------------------------------------------------------------
// Kernel 1: persistent per-batch recurrence.
// grid = 8 (one WG per batch), 256 threads = 8 waves.
// Wave w owns a 4x4 tile block: M tiles m0..m0+3 (W dim), N tiles n0..n0+3 (Cout).
// ---------------------------------------------------------------------------
extern __shared__ __align__(16) unsigned char smem_raw[];

__global__ void __launch_bounds__(256, 1)
msgpass_kernel(const float* __restrict__ x,
               float* __restrict__ out,
               const unsigned short* __restrict__ wbf) {
    const int b    = blockIdx.x;
    const int tid  = threadIdx.x;
    const int wv   = tid >> 5;
    const int lane = tid & 31;
    const int hf   = lane >> 4;     // 0 or 1 (half-wave)
    const int mrow = lane & 15;     // M index within tile (A-matrix lane role)
    const int ncol = lane & 15;     // N index within tile (B/C/D lane role)

    const int m0 = (wv & 3) * 4;    // M tiles (of 16) this wave owns
    const int n0 = (wv >> 2) * 4;   // N tiles (of 8) this wave owns

    // --- Stage bf16 weights into LDS ------------------------------------
#if __has_builtin(__builtin_amdgcn_tensor_load_to_lds)
    if (wv == 0) {
        // TDM descriptor: 1-D copy, data_size=8B, 36864 elements = 288 KB.
        unsigned long long ga = (unsigned long long)wbf;
        uint4v g0;
        g0.x = 1u;                                  // count=1, user mode
        g0.y = 0u;                                  // lds_addr = 0
        g0.z = (unsigned int)(ga & 0xFFFFFFFFull);  // global_addr[31:0]
        g0.w = (unsigned int)((ga >> 32) & 0x01FFFFFFull) | (2u << 30); // addr[56:32] | type=2
        int8v g1;
        g1[0] = (int)(3u << 16);                    // data_size = 3 (8 bytes)
        g1[1] = (int)0x90000000u;                   // tensor_dim0 lo16 (0x9000) << 16
        g1[2] = (int)0x00010000u;                   // tensor_dim0 hi=0 | tensor_dim1 lo16 = 1
        g1[3] = (int)0x90000000u;                   // tensor_dim1 hi=0 | tile_dim0 = 0x9000
        g1[4] = 1;                                  // tile_dim1 = 1, tile_dim2 = 0
        g1[5] = 36864;                              // tensor_dim0_stride lo32
        g1[6] = 0;
        g1[7] = 0;
        int4v z4 = {0, 0, 0, 0};
        int8v z8 = {0, 0, 0, 0, 0, 0, 0, 0};
        __builtin_amdgcn_tensor_load_to_lds(g0, g1, z4, z4, z8, 0);
        __builtin_amdgcn_s_wait_tensorcnt(0);
    }
#endif
    __syncthreads();
    // Deterministic fallback/verify copy (writes identical bytes).
    {
        const uint4* src = (const uint4*)wbf;
        uint4*       dst = (uint4*)smem_raw;
        for (int t = tid; t < WBYTES / 16; t += 256) dst[t] = src[t];
    }
    __syncthreads();

    // --- Row 0: pure copy x[b,0] -> out[b,0] ----------------------------
    {
        const size_t base = (size_t)b * HH * WW * CC;
        const float4* src = (const float4*)(x + base);
        float4*       dst = (float4*)(out + base);
        for (int t = tid; t < (WW * CC) / 4; t += 256) dst[t] = src[t];
    }

    // --- Recurrence over rows -------------------------------------------
    for (int i = 1; i < HH; ++i) {
        // make this WG's stores to row i-1 visible, invalidate WGP cache
        __threadfence();
        __syncthreads();
        __threadfence();

        const float* prev = out + ((size_t)b * HH + (i - 1)) * WW * CC;
        const float* xrow = x   + ((size_t)b * HH + i) * WW * CC;
        float*       orow = out + ((size_t)b * HH + i) * WW * CC;

        v8f acc[16];
        const v8f vzero = {0.f, 0.f, 0.f, 0.f, 0.f, 0.f, 0.f, 0.f};
#pragma unroll
        for (int t = 0; t < 16; ++t) acc[t] = vzero;

        for (int kc = 0; kc < 4; ++kc) {
            for (int tap = 0; tap < KK; ++tap) {
                // B fragments for this wave's 4 N tiles (from LDS)
                v16bf bf[4];
#pragma unroll
                for (int nj = 0; nj < 4; ++nj) {
                    const int frag = ((tap * 4 + kc) * 8 + (n0 + nj));
                    bf[nj] = *(const v16bf*)(smem_raw + (frag << 10) + (lane << 5));
                }
#pragma unroll
                for (int mi = 0; mi < 4; ++mi) {
                    // A fragment: rows of prev row, fp32 -> bf16 (packed perm)
                    const int w     = (m0 + mi) * 16 + mrow + tap - PAD;
                    const bool ok   = (w >= 0) && (w < WW);
                    const int  wS   = ok ? w : 0;
                    const int  cib  = kc * 32 + hf * 16;
                    const float4* p = (const float4*)(prev + (size_t)wS * CC + cib);
                    const float4* q = (const float4*)(prev + (size_t)wS * CC + cib + 8);
                    float4 a0 = p[0], a1 = p[1], b0 = q[0], b1 = q[1];
                    if (!ok) {
                        a0 = make_float4(0.f, 0.f, 0.f, 0.f);
                        a1 = a0; b0 = a0; b1 = a0;
                    }
                    union { v16bf v; unsigned int d[8]; } af;
                    af.d[0] = pack_bf16_trunc(a0.x, a0.y);
                    af.d[1] = pack_bf16_trunc(a0.z, a0.w);
                    af.d[2] = pack_bf16_trunc(a1.x, a1.y);
                    af.d[3] = pack_bf16_trunc(a1.z, a1.w);
                    af.d[4] = pack_bf16_trunc(b0.x, b0.y);
                    af.d[5] = pack_bf16_trunc(b0.z, b0.w);
                    af.d[6] = pack_bf16_trunc(b1.x, b1.y);
                    af.d[7] = pack_bf16_trunc(b1.z, b1.w);
#pragma unroll
                    for (int nj = 0; nj < 4; ++nj) {
                        acc[mi * 4 + nj] = __builtin_amdgcn_wmma_f32_16x16x32_bf16(
                            false, af.v, false, bf[nj], (short)0,
                            acc[mi * 4 + nj], false, false);
                    }
                }
            }
        }

        // Epilogue: out = x + relu(conv). C/D layout: VGPR r -> M = r + 8*half,
        // lane%16 -> N.
#pragma unroll
        for (int mi = 0; mi < 4; ++mi) {
            const int tw0 = (m0 + mi) * 16;
#pragma unroll
            for (int nj = 0; nj < 4; ++nj) {
                const int c = (n0 + nj) * 16 + ncol;
#pragma unroll
                for (int r = 0; r < 8; ++r) {
                    const int w = tw0 + r + hf * 8;
                    const size_t idx = (size_t)w * CC + c;
                    float v = acc[mi * 4 + nj][r];
                    v = v > 0.f ? v : 0.f;
                    orow[idx] = xrow[idx] + v;
                }
            }
        }
    }
}

// ---------------------------------------------------------------------------
extern "C" void kernel_launch(void* const* d_in, const int* in_sizes, int n_in,
                              void* d_out, int out_size, void* d_ws, size_t ws_size,
                              hipStream_t stream) {
    const float* x  = (const float*)d_in[0];   // (8,128,256,128) fp32
    const float* Wf = (const float*)d_in[1];   // (9,128,128) fp32
    float* out      = (float*)d_out;           // (8,128,256,128) fp32
    unsigned short* wbf = (unsigned short*)d_ws;

    // 9*128*128 = 147456 bf16 fragment elements
    wcvt_kernel<<<576, 256, 0, stream>>>(Wf, wbf);

    (void)hipFuncSetAttribute((const void*)msgpass_kernel,
                              hipFuncAttributeMaxDynamicSharedMemorySize, WBYTES);
    msgpass_kernel<<<BB, 256, WBYTES, stream>>>(x, out, wbf);
}
